// AlignmentNet_2164663517268
// MI455X (gfx1250) — compile-verified, matrix operand order checked by reading, not run on hardware
//
#include <hip/hip_runtime.h>
#include <hip/hip_bf16.h>

// ---------------------------------------------------------------------------
// Problem constants (match reference)
// ---------------------------------------------------------------------------
#define BN   4
#define CCH  64
#define HH   128
#define WW   128
#define GG   4
#define KKT  9           // K*K taps
#define OFFCH 72         // 2*K*K*G
#define STRIP 16         // pixels per block strip
#define STRIPS (WW / STRIP)

typedef __attribute__((ext_vector_type(16))) __bf16        v16bf;
typedef __attribute__((ext_vector_type(16))) short         v16s;
typedef __attribute__((ext_vector_type(8)))  float         v8f;
typedef __attribute__((ext_vector_type(4)))  unsigned int  v4u;
typedef __attribute__((ext_vector_type(8)))  int           v8i;
typedef __attribute__((ext_vector_type(4)))  int           v4i;

#if defined(__has_builtin)
#if __has_builtin(__builtin_amdgcn_tensor_load_to_lds) && \
    __has_builtin(__builtin_amdgcn_s_wait_tensorcnt)
#define HAVE_TDM 1
#endif
#endif
#ifndef HAVE_TDM
#define HAVE_TDM 0
#endif

// fp32 -> bf16 (round to nearest even), bit-level so we never rely on __bf16 math
__device__ __forceinline__ unsigned short f2bf(float f) {
    unsigned int u = __builtin_bit_cast(unsigned int, f);
    u += 0x7FFFu + ((u >> 16) & 1u);
    return (unsigned short)(u >> 16);
}

__device__ __forceinline__ int clampi(int v, int lo, int hi) {
    return v < lo ? lo : (v > hi ? hi : v);
}

// ISA 16-bit fragment K map: element i of the per-lane v16, with lane half h:
//   k = (i&7) + ((i>>3)<<4) + h*8        (covers K=0..31 across both halves)
__device__ __forceinline__ int frag_k(int i, int half) {
    return (i & 7) + ((i >> 3) << 4) + half * 8;
}

#if HAVE_TDM
// ---------------------------------------------------------------------------
// Tensor Data Mover: 3D tile load (X=tile0 f32 elems, Y=tile1 rows @stride0,
// Z=tile2 planes @stride1) from global into contiguous LDS at lds_byte.
// D# built per CDNA5 ISA (group0..3).
// This toolchain exposes the 6-arg builtin:
//   (uint32x4 g0, int32x8 g1, int32x4 g2, int32x4 g3, int32x8 extra, i32 cpol)
// The trailing int32x8 is zeroed (single descriptor, count=1 in group0).
// Tensor dims are set equal to tile dims, so no OOB clipping occurs (caller
// guarantees the tile is fully inside the image).
// ---------------------------------------------------------------------------
__device__ __forceinline__ void tdm_load_3d(const float* gsrc, unsigned lds_byte,
                                            int tile0, int tile1, int tile2,
                                            long long stride0, long long stride1)
{
    unsigned long long ga = (unsigned long long)(__SIZE_TYPE__)gsrc;
    v4u g0;
    g0[0] = 1u;                                         // count=1 (valid user D#)
    g0[1] = lds_byte;                                   // lds_addr
    g0[2] = (unsigned)(ga & 0xFFFFFFFFu);               // global_addr[31:0]
    g0[3] = (unsigned)((ga >> 32) & 0x1FFFFFFu)         // global_addr[56:32]
          | (2u << 30);                                 // type=2 ("image")
    v8i g1;
    g1[0] = (int)(2u << 16);                            // wg_mask=0, data_size=4B
    g1[1] = (int)((unsigned)tile0 << 16);               // tensor_dim0[15:0]
    g1[2] = (int)((unsigned)tile1 << 16);               // tensor_dim1[15:0]
    g1[3] = (int)((unsigned)tile0 << 16);               // tile_dim0
    g1[4] = (int)((unsigned)tile1 | ((unsigned)tile2 << 16)); // tile_dim1 | tile_dim2
    g1[5] = (int)(unsigned)(stride0 & 0xFFFFFFFFll);    // tensor_dim0_stride[31:0]
    g1[6] = (int)(((unsigned)(stride0 >> 32) & 0xFFFFu) // stride0[47:32]
          | (((unsigned)stride1 & 0xFFFFu) << 16));     // stride1[15:0]
    g1[7] = (int)(unsigned)((stride1 >> 16) & 0xFFFFFFFFll); // stride1[47:16]
    v4i g2;
    g2[0] = tile2;                                      // tensor_dim2
    g2[1] = 1;                                          // tensor_dim3 (unused)
    g2[2] = 0;                                          // tensor_dim2_stride[31:0]
    g2[3] = 0;                                          // stride2 hi | tile_dim3=0
    v4i g3 = {0, 0, 0, 0};                              // dims 3/4 unused
    v8i g4 = {0, 0, 0, 0, 0, 0, 0, 0};                  // no second descriptor
    __builtin_amdgcn_tensor_load_to_lds(g0, g1, g2, g3, g4, 0);
}
#endif

// ---------------------------------------------------------------------------
// Weight packing: OIHW f32 -> bf16 WMMA A-fragments.
// Layout: [coutTile][ktile][lane][16 bf16]  (one 32B contiguous chunk per lane)
// ---------------------------------------------------------------------------
__global__ void pack_conv_w(const float* __restrict__ w,
                            unsigned short* __restrict__ wp,
                            int Cin, int Cout, int coutTiles, int KT)
{
    int idx = blockIdx.x * blockDim.x + threadIdx.x;
    int total = coutTiles * KT * 32 * 16;
    if (idx >= total) return;
    int i    = idx & 15;
    int lane = (idx >> 4) & 31;
    int rest = idx >> 9;            // 32*16 = 512 = 2^9
    int kt   = rest % KT;
    int ct   = rest / KT;
    int m    = lane & 15;
    int half = lane >> 4;
    int k    = kt * 32 + frag_k(i, half);
    int cout = ct * 16 + m;
    float v = 0.f;
    if (cout < Cout && k < Cin * 9) {
        int cin = k / 9;
        int tap = k - cin * 9;
        v = w[(cout * Cin + cin) * 9 + tap];
    }
    wp[idx] = f2bf(v);
}

// Deform-conv weights (G=4, Coutg=16, Cg=16, 9 taps): per group an A matrix
// M=16 x K=144 (padded to 160 -> 5 ktiles). Layout [g][kt(5)][lane][16].
__global__ void pack_dconv_w(const float* __restrict__ w,
                             unsigned short* __restrict__ wp)
{
    int idx = blockIdx.x * blockDim.x + threadIdx.x;
    if (idx >= GG * 5 * 32 * 16) return;
    int i    = idx & 15;
    int lane = (idx >> 4) & 31;
    int rest = idx >> 9;
    int kt   = rest % 5;
    int g    = rest / 5;
    int m    = lane & 15;
    int half = lane >> 4;
    int k    = kt * 32 + frag_k(i, half);
    float v = 0.f;
    if (k < 16 * KKT) {
        int c   = k / KKT;
        int tap = k - c * KKT;
        v = w[((g * 16 + m) * 16 + c) * KKT + tap];
    }
    wp[idx] = f2bf(v);
}

// ---------------------------------------------------------------------------
// 3x3 conv (pad=1, dil=1) as implicit GEMM with WMMA bf16.
//   grid  : BN * HH * STRIPS blocks, block = 32*NW threads (1 wave / Cout tile)
//   CIN   : 64 or 128 (128 = concat of in0,in1 64-channel tensors)
//   GEMM  : M = 16 couts (per wave), N = 16 pixels, K = CIN*9 (multiple of 32)
// Interior blocks stage the f32 halo patch via the Tensor Data Mover (one
// 18x3x64 tile per 64-ch tensor); border blocks fall back to manual staging.
// ---------------------------------------------------------------------------
template <int CIN, int NW>
__global__ void conv3x3_wmma(const float* __restrict__ in0,
                             const float* __restrict__ in1,
                             const float* __restrict__ bias,
                             const unsigned short* __restrict__ wpack,
                             float* __restrict__ out,
                             int Cout)
{
    constexpr int KT   = (CIN * 9) / 32;    // 36 (CIN=128) or 18 (CIN=64)
    constexpr int LDSW = 18;                // contiguous rows (TDM-compatible)
    __shared__ float patch[CIN * 3 * LDSW]; // f32 input halo patch

    const int tid  = threadIdx.x;
    const int lane = tid & 31;
    const int wave = tid >> 5;
    const int bid  = blockIdx.x;
    const int b    = bid / (HH * STRIPS);
    const int rem  = bid % (HH * STRIPS);
    const int y    = rem / STRIPS;
    const int x0   = (rem % STRIPS) * STRIP;

    // Prefetch this wave's packed weight fragments while we stage the patch.
    __builtin_prefetch(wpack + (size_t)wave * KT * 512, 0, 0);

    const bool interior = (y > 0) && (y < HH - 1) && (x0 > 0) && (x0 + STRIP < WW);

#if HAVE_TDM
    if (interior) {
        if (tid < 32) {     // wave 0 issues the TDM descriptors (EXEC-independent)
            unsigned ldsb = (unsigned)(__SIZE_TYPE__)(void*)&patch[0];
            const float* src0 = in0 + (((size_t)b * 64) * HH + (y - 1)) * WW + (x0 - 1);
            tdm_load_3d(src0, ldsb, LDSW, 3, 64, WW, (long long)HH * WW);
            if (CIN == 128) {
                const float* src1 = in1 + (((size_t)b * 64) * HH + (y - 1)) * WW + (x0 - 1);
                tdm_load_3d(src1, ldsb + 64u * 3u * LDSW * 4u, LDSW, 3, 64,
                            WW, (long long)HH * WW);
            }
            __builtin_amdgcn_s_wait_tensorcnt(0);
        }
    } else
#endif
    {
        // Manual staging with zero halo (also the no-TDM fallback for all blocks).
        const int total = CIN * 3 * LDSW;
        for (int idx = tid; idx < total; idx += 32 * NW) {
            int c  = idx / (3 * LDSW);
            int r2 = idx - c * (3 * LDSW);
            int r  = r2 / LDSW;
            int cc = r2 - r * LDSW;
            int iy = y - 1 + r;
            int ix = x0 - 1 + cc;
            float v = 0.f;
            if (iy >= 0 && iy < HH && ix >= 0 && ix < WW) {
                const float* src = (c < 64) ? in0 : in1;
                int c2 = (c < 64) ? c : (c - 64);
                v = src[((b * 64 + c2) * HH + iy) * WW + ix];
            }
            patch[idx] = v;
        }
    }
    __syncthreads();

    const int half = lane >> 4;
    const int n    = lane & 15;

    v8f acc = {};
    const v16s* pA = (const v16s*)wpack + wave * KT * 32;
#pragma unroll 2
    for (int kt = 0; kt < KT; ++kt) {
        v16s as = pA[kt * 32 + lane];       // one 32B fragment load per lane
        v16s bs;
#pragma unroll
        for (int i = 0; i < 16; ++i) {
            int k   = kt * 32 + frag_k(i, half);   // k < CIN*9 always (exact)
            int c   = k / 9;
            int tap = k - c * 9;
            int kh  = tap / 3;
            int kw  = tap - kh * 3;
            bs[i] = (short)f2bf(patch[(c * 3 + kh) * LDSW + (n + kw)]);
        }
        v16bf av = __builtin_bit_cast(v16bf, as);
        v16bf bv = __builtin_bit_cast(v16bf, bs);
        acc = __builtin_amdgcn_wmma_f32_16x16x32_bf16(
                  false, av, false, bv, (short)0, acc, false, false);
    }

    // D layout: VGPR r -> M = half*8 + r, N = lane%16
#pragma unroll
    for (int r = 0; r < 8; ++r) {
        int cout = wave * 16 + half * 8 + r;
        if (cout < Cout) {
            out[((b * Cout + cout) * HH + y) * WW + x0 + n] = acc[r] + bias[cout];
        }
    }
}

// ---------------------------------------------------------------------------
// Grouped deformable conv (G=4, Cg=Coutg=16, K=3, PAD=2, DIL=2) with WMMA.
//   block = 128 threads (4 waves, one per group); grid = BN*HH*STRIPS.
//   Per group: bilinear-sample (Cg x 9 x 16pix) into bf16 LDS (K padded 144->160),
//   then GEMM M=16 (Coutg) x N=16 (pixels) x K=160 via 5 WMMAs.
// ---------------------------------------------------------------------------
__global__ void deform_conv_wmma(const float* __restrict__ x,      // (B,64,H,W)
                                 const float* __restrict__ off,    // (B,72,H,W)
                                 const unsigned short* __restrict__ wpack, // [4][5][32][16]
                                 const float* __restrict__ bias,
                                 float* __restrict__ out)           // (B,64,H,W)
{
    __shared__ unsigned short smp[GG][160 * 16];   // 20 KB sampled B tiles

    const int tid  = threadIdx.x;
    const int lane = tid & 31;
    const int g    = tid >> 5;
    const int bid  = blockIdx.x;
    const int b    = bid / (HH * STRIPS);
    const int rem  = bid % (HH * STRIPS);
    const int y    = rem / STRIPS;
    const int x0   = (rem % STRIPS) * STRIP;

    // Bilinear sampling: 9 taps x 16 pixels per group, 16 channels each.
    for (int pr = lane; pr < KKT * STRIP; pr += 32) {
        int kk = pr / STRIP;        // tap 0..8
        int n  = pr & (STRIP - 1);  // pixel in strip
        int xx = x0 + n;
        int offc = (g * KKT + kk) * 2;
        float dy = off[((b * OFFCH + offc    ) * HH + y) * WW + xx];
        float dx = off[((b * OFFCH + offc + 1) * HH + y) * WW + xx];
        float py = dy + (float)(y  - 2 + 2 * (kk / 3));
        float px = dx + (float)(xx - 2 + 2 * (kk % 3));
        float fy = floorf(py), fx = floorf(px);
        int iy0 = (int)fy, ix0 = (int)fx;
        int iy1 = iy0 + 1, ix1 = ix0 + 1;
        float ly = py - fy, lx = px - fx;
        float w00 = (1.f - ly) * (1.f - lx);
        float w01 = (1.f - ly) * lx;
        float w10 = ly * (1.f - lx);
        float w11 = ly * lx;
        bool vy0 = (iy0 >= 0) && (iy0 < HH), vy1 = (iy1 >= 0) && (iy1 < HH);
        bool vx0 = (ix0 >= 0) && (ix0 < WW), vx1 = (ix1 >= 0) && (ix1 < WW);
        if (!(vy0 && vx0)) w00 = 0.f;
        if (!(vy0 && vx1)) w01 = 0.f;
        if (!(vy1 && vx0)) w10 = 0.f;
        if (!(vy1 && vx1)) w11 = 0.f;
        int cy0 = clampi(iy0, 0, HH - 1), cy1 = clampi(iy1, 0, HH - 1);
        int cx0 = clampi(ix0, 0, WW - 1), cx1 = clampi(ix1, 0, WW - 1);
        const float* base = x + ((size_t)b * 64 + g * 16) * HH * WW;
#pragma unroll 4
        for (int c = 0; c < 16; ++c) {
            const float* img = base + (size_t)c * HH * WW;
            float s = w00 * img[cy0 * WW + cx0] + w01 * img[cy0 * WW + cx1]
                    + w10 * img[cy1 * WW + cx0] + w11 * img[cy1 * WW + cx1];
            smp[g][(c * KKT + kk) * 16 + n] = f2bf(s);
        }
    }
    // Zero the K padding rows 144..159.
    for (int t = lane; t < 16 * 16; t += 32) {
        smp[g][144 * 16 + t] = 0;
    }
    __syncthreads();

    const int half = lane >> 4;
    const int n    = lane & 15;

    v8f acc = {};
    const v16s* pA = (const v16s*)wpack + g * 5 * 32;
#pragma unroll
    for (int kt = 0; kt < 5; ++kt) {
        v16s as = pA[kt * 32 + lane];
        v16s bs;
#pragma unroll
        for (int i = 0; i < 16; ++i) {
            int k = kt * 32 + frag_k(i, half);
            bs[i] = (short)smp[g][k * 16 + n];
        }
        v16bf av = __builtin_bit_cast(v16bf, as);
        v16bf bv = __builtin_bit_cast(v16bf, bs);
        acc = __builtin_amdgcn_wmma_f32_16x16x32_bf16(
                  false, av, false, bv, (short)0, acc, false, false);
    }

#pragma unroll
    for (int r = 0; r < 8; ++r) {
        int cout = g * 16 + half * 8 + r;
        out[((b * 64 + cout) * HH + y) * WW + x0 + n] = acc[r] + bias[cout];
    }
}

// ---------------------------------------------------------------------------
// Host orchestration
// ---------------------------------------------------------------------------
extern "C" void kernel_launch(void* const* d_in, const int* in_sizes, int n_in,
                              void* d_out, int out_size, void* d_ws, size_t ws_size,
                              hipStream_t stream)
{
    (void)in_sizes; (void)n_in; (void)out_size; (void)ws_size;

    const float* Fref  = (const float*)d_in[0];
    const float* Fmov1 = (const float*)d_in[1];
    const float* Fmov2 = (const float*)d_in[2];
    const float* cr_w  = (const float*)d_in[3];
    const float* cr_b  = (const float*)d_in[4];
    const float* off_w[4] = { (const float*)d_in[5],  (const float*)d_in[7],
                              (const float*)d_in[9],  (const float*)d_in[11] };
    const float* off_b[4] = { (const float*)d_in[6],  (const float*)d_in[8],
                              (const float*)d_in[10], (const float*)d_in[12] };
    const float* dw[4]    = { (const float*)d_in[13], (const float*)d_in[15],
                              (const float*)d_in[17], (const float*)d_in[19] };
    const float* db[4]    = { (const float*)d_in[14], (const float*)d_in[16],
                              (const float*)d_in[18], (const float*)d_in[20] };

    // Workspace carve-up (~53 MB total; everything 32B-aligned).
    const size_t feaElems = (size_t)BN * 64 * HH * WW;      // 4,194,304
    const size_t offElems = (size_t)BN * OFFCH * HH * WW;   // 4,718,592
    char* p = (char*)d_ws;
    float* feaA = (float*)p;            p += feaElems * 4;
    float* feaB = (float*)p;            p += feaElems * 4;
    float* offb = (float*)p;            p += offElems * 4;
    unsigned short* wp_cr = (unsigned short*)p;  p += (size_t)4 * 36 * 512 * 2;
    unsigned short* wp_off[4];
    for (int i = 0; i < 4; ++i) { wp_off[i] = (unsigned short*)p; p += (size_t)5 * 18 * 512 * 2; }
    unsigned short* wp_d[4];
    for (int i = 0; i < 4; ++i) { wp_d[i]   = (unsigned short*)p; p += (size_t)GG * 5 * 512 * 2; }

    // Pack all weights into WMMA A-fragment order (bf16).
    {
        int tot = 4 * 36 * 512;
        pack_conv_w<<<(tot + 255) / 256, 256, 0, stream>>>(cr_w, wp_cr, 128, 64, 4, 36);
    }
    for (int i = 0; i < 4; ++i) {
        int tot = 5 * 18 * 512;
        pack_conv_w<<<(tot + 255) / 256, 256, 0, stream>>>(off_w[i], wp_off[i], 64, OFFCH, 5, 18);
    }
    for (int i = 0; i < 4; ++i) {
        int tot = GG * 5 * 512;
        pack_dconv_w<<<(tot + 255) / 256, 256, 0, stream>>>(dw[i], wp_d[i]);
    }

    const int blocks = BN * HH * STRIPS;   // 4096
    float* out = (float*)d_out;

    // align(fr, fm) -> dst
    auto align = [&](const float* fr, const float* fm, float* dst) {
        // fea0 = conv(concat(fr,fm)) -> feaA
        conv3x3_wmma<128, 4><<<blocks, 128, 0, stream>>>(fr, fm, cr_b, wp_cr, feaA, 64);
        // off1 = conv(feaA); fea1 = dconv(feaA, off1) -> feaB
        conv3x3_wmma<64, 5><<<blocks, 160, 0, stream>>>(feaA, feaA, off_b[0], wp_off[0], offb, OFFCH);
        deform_conv_wmma<<<blocks, 128, 0, stream>>>(feaA, offb, wp_d[0], db[0], feaB);
        // off2 = conv(feaB); fea2 = dconv(feaB, off2) -> feaA
        conv3x3_wmma<64, 5><<<blocks, 160, 0, stream>>>(feaB, feaB, off_b[1], wp_off[1], offb, OFFCH);
        deform_conv_wmma<<<blocks, 128, 0, stream>>>(feaB, offb, wp_d[1], db[1], feaA);
        // off3 = conv(feaA); fea3 = dconv(fm, off3) -> feaB
        conv3x3_wmma<64, 5><<<blocks, 160, 0, stream>>>(feaA, feaA, off_b[2], wp_off[2], offb, OFFCH);
        deform_conv_wmma<<<blocks, 128, 0, stream>>>(fm, offb, wp_d[2], db[2], feaB);
        // off4 = conv(feaB); out = dconv(feaB, off4) -> dst
        conv3x3_wmma<64, 5><<<blocks, 160, 0, stream>>>(feaB, feaB, off_b[3], wp_off[3], offb, OFFCH);
        deform_conv_wmma<<<blocks, 128, 0, stream>>>(feaB, offb, wp_d[3], db[3], dst);
    };

    align(Fref, Fmov1, out);
    align(Fref, Fmov2, out + feaElems);
}
